// MultiHeadAttention_11269994185361
// MI455X (gfx1250) — compile-verified, matrix operand order checked by reading, not run on hardware
//
#include <hip/hip_runtime.h>
#include <hip/hip_bf16.h>

// ---------------- types ----------------
typedef __bf16 bf16_t;
typedef __attribute__((ext_vector_type(4)))  __bf16 bfx4;
typedef __attribute__((ext_vector_type(8)))  __bf16 bfx8;
typedef __attribute__((ext_vector_type(16))) __bf16 bfx16;
typedef __attribute__((ext_vector_type(8)))  float  fx8;
typedef __attribute__((ext_vector_type(4)))  int    v4i;

#define D_MODEL 1024
#define NHEADS  16
#define DK      64
#define SEQ     2048
#define BATCH   4
#define NTOK    (BATCH*SEQ)   // 8192

// ---- gfx1250 async copy (global -> LDS, ASYNCcnt-tracked) with fallback ----
#if __has_builtin(__builtin_amdgcn_global_load_async_to_lds_b128) && \
    __has_builtin(__builtin_amdgcn_s_wait_asynccnt)
#define HAS_ASYNC 1
#else
#define HAS_ASYNC 0
#endif

#if HAS_ASYNC
// signature (probe-confirmed by diagnostics): (v4i AS1* src, v4i AS3* dst, Ii off, Ii cpol)
#define KV_CPY(dst, src)                                                   \
  __builtin_amdgcn_global_load_async_to_lds_b128(                          \
      (__attribute__((address_space(1))) v4i*)(src),                       \
      (__attribute__((address_space(3))) v4i*)(dst), 0, 0)
#define WAIT_ASYNC() __builtin_amdgcn_s_wait_asynccnt(0)
#else
#define KV_CPY(dst, src) do { *(bfx8*)(dst) = *(const bfx8*)(src); } while (0)
#define WAIT_ASYNC() do {} while (0)
#endif

// Load a 16-bf16 WMMA A/B fragment slice for this lane.
// Per ISA 7.12.2 (16-bit A 16x32): lane<16 holds K = kbase+{0..7, 16..23},
// lane>=16 holds K = kbase+8+{0..7, 16..23}. Caller passes p already offset
// by (half*8); the two runs are p[0..7] and p[16..23] -> two b128 loads.
template <typename P>
__device__ __forceinline__ bfx16 load_frag(P p) {
  bfx8 lo = *(const bfx8*)(p);
  bfx8 hi = *(const bfx8*)(p + 16);
  return __builtin_shufflevector(lo, hi, 0,1,2,3,4,5,6,7,8,9,10,11,12,13,14,15);
}

__device__ __forceinline__ fx8 wmma_bf16(bfx16 a, bfx16 b, fx8 c) {
  // (neg_a, A, neg_b, B, c_mod, C, reuse_a, reuse_b)
  return __builtin_amdgcn_wmma_f32_16x16x32_bf16(false, a, false, b, (short)0, c, false, false);
}

// ---------------- conversion kernels ----------------
__global__ void cvt_bf16_4(const float* __restrict__ src, bf16_t* __restrict__ dst, int n4) {
  int i = blockIdx.x * 256 + threadIdx.x;
  if (i < n4) {
    const float4 f = ((const float4*)src)[i];
    bfx4 o = { (__bf16)f.x, (__bf16)f.y, (__bf16)f.z, (__bf16)f.w };
    ((bfx4*)dst)[i] = o;
  }
}

// dst[n*K + k] = src[k*N + n]  (weight transpose: [K,N] fp32 -> [N,K] bf16)
__global__ void cvt_tr(const float* __restrict__ src, bf16_t* __restrict__ dst, int K, int N) {
  int i = blockIdx.x * 256 + threadIdx.x;
  if (i < K * N) {
    int n = i / K, k = i % K;
    dst[i] = (bf16_t)src[(size_t)k * N + n];
  }
}

// ---------------- GEMM: C[M,N] = A[M,K] * Bt[N,K]^T + bias ----------------
// mode 0: out bf16 scattered to [B,H,S,DK]        (q,k projections)
// mode 1: out bf16 scattered to [B,H,DK,S]        (v projection, transposed)
// mode 2: out fp32 row-major [M,N]                (output projection)
__global__ __launch_bounds__(32) void gemm_bf16_k(
    const bf16_t* __restrict__ A, const bf16_t* __restrict__ Bt,
    const float* __restrict__ bias,
    bf16_t* __restrict__ outB, float* __restrict__ outF,
    int M, int N, int K, int mode)
{
  const int lane = threadIdx.x & 31;
  const int l16  = lane & 15;
  const int hf   = lane >> 4;
  const int m0   = blockIdx.x * 64;
  const int n0   = blockIdx.y * 64;

  fx8 acc[4][4];
#pragma unroll
  for (int i = 0; i < 4; ++i)
#pragma unroll
    for (int j = 0; j < 4; ++j) acc[i][j] = (fx8){};

  const bf16_t* aBase[4];
  const bf16_t* bBase[4];
#pragma unroll
  for (int i = 0; i < 4; ++i) aBase[i] = A  + (size_t)(m0 + i*16 + l16) * K + hf*8;
#pragma unroll
  for (int j = 0; j < 4; ++j) bBase[j] = Bt + (size_t)(n0 + j*16 + l16) * K + hf*8;

  for (int k0 = 0; k0 < K; k0 += 32) {
    bfx16 af[4], bfr[4];
#pragma unroll
    for (int i = 0; i < 4; ++i) af[i]  = load_frag(aBase[i] + k0);
#pragma unroll
    for (int j = 0; j < 4; ++j) bfr[j] = load_frag(bBase[j] + k0);
    if (k0 + 32 < K) {               // global_prefetch next K-slab
      __builtin_prefetch(aBase[0] + k0 + 32, 0, 3);
      __builtin_prefetch(bBase[0] + k0 + 32, 0, 3);
    }
#pragma unroll
    for (int i = 0; i < 4; ++i)
#pragma unroll
      for (int j = 0; j < 4; ++j)
        acc[i][j] = wmma_bf16(af[i], bfr[j], acc[i][j]);
  }

  // epilogue: C/D layout -> lane l holds col n=l16, rows m = v + hf*8
#pragma unroll
  for (int j = 0; j < 4; ++j) {
    int c = n0 + j*16 + l16;
    float bv = bias[c];
#pragma unroll
    for (int i = 0; i < 4; ++i) {
#pragma unroll
      for (int v = 0; v < 8; ++v) {
        int r = m0 + i*16 + v + hf*8;
        float val = acc[i][j][v] + bv;
        if (mode == 2) {
          outF[(size_t)r * N + c] = val;
        } else {
          int b = r >> 11, s = r & 2047;      // S = 2048
          int h = c >> 6,  dk = c & 63;       // DK = 64
          size_t idx = (mode == 0)
            ? (((size_t)(b*NHEADS + h) * SEQ + s) * DK + dk)
            : (((size_t)(b*NHEADS + h) * DK + dk) * SEQ + s);
          outB[idx] = (bf16_t)val;
        }
      }
    }
  }
}

// ---------------- flash attention ----------------
// 128-thread block = 4 waves, all on the same (b,h); each wave owns one
// 16-query tile. K/V blocks (64 keys) are staged into LDS cooperatively
// (async copy, double-buffered) and shared by all 4 waves -> 4x less L2
// traffic for the streaming operands.
// qp,kp: [B,H,S,DK] bf16 ; vT: [B,H,DK,S] bf16 ; attnC: [NTOK, D_MODEL] bf16
__global__ __launch_bounds__(128) void flash_k(
    const bf16_t* __restrict__ qp, const bf16_t* __restrict__ kp,
    const bf16_t* __restrict__ vT, bf16_t* __restrict__ attnC)
{
  // ldsK[buf]: 64 keys x 64 d (row-major, identical to global layout)
  // ldsV[buf]: 64 d x 64 keys (row-major, identical to vT block layout)
  __shared__ __align__(16) bf16_t ldsK[2][64 * DK];
  __shared__ __align__(16) bf16_t ldsV[2][DK * 64];
  __shared__ __align__(16) bf16_t lp[4][16 * 64];   // per-wave private P tile

  const int tid  = threadIdx.x;
  const int wv   = tid >> 5;
  const int lane = tid & 31;
  const int l16  = lane & 15;
  const int hf   = lane >> 4;
  const int bh   = blockIdx.y;            // 0..63
  const int b    = bh >> 4, h = bh & 15;
  const int q0   = (blockIdx.x * 4 + wv) * 16;

  const bf16_t* qb  = qp + (size_t)bh * SEQ * DK;
  const bf16_t* kbp = kp + (size_t)bh * SEQ * DK;
  const bf16_t* vb  = vT + (size_t)bh * DK * SEQ;
  bf16_t* lpw = &lp[wv][0];

  // q tile as two A fragments (16 x 32 each), kept in registers for all keys
  bfx16 aq0 = load_frag(qb + (size_t)(q0 + l16) * DK +  0 + hf*8);
  bfx16 aq1 = load_frag(qb + (size_t)(q0 + l16) * DK + 32 + hf*8);

  fx8 accO[4];
#pragma unroll
  for (int nt = 0; nt < 4; ++nt) accO[nt] = (fx8){};
  float mrow[8], lrow[8];
#pragma unroll
  for (int v = 0; v < 8; ++v) { mrow[v] = -1e30f; lrow[v] = 0.f; }

  // cooperative stage of one 64-key K/V block into LDS buffer `buf`
  auto stage = [&](int buf, int kb0) {
    // K block: 64*64 bf16 = 8KB, fully contiguous in global
    const bf16_t* kSrc = kbp + (size_t)kb0 * DK;
#pragma unroll
    for (int c = 0; c < 4; ++c) {
      int idx = tid + c * 128;                 // 512 chunks of 16B
      KV_CPY(&ldsK[buf][idx * 8], kSrc + idx * 8);
    }
    // V block: 64 rows (d), each 64 keys, row stride SEQ in global
#pragma unroll
    for (int c = 0; c < 4; ++c) {
      int idx = tid + c * 128;
      int d = idx >> 3, col = (idx & 7) * 8;
      KV_CPY(&ldsV[buf][d * 64 + col], vb + (size_t)d * SEQ + kb0 + col);
    }
  };

  stage(0, 0);
  WAIT_ASYNC();
  __syncthreads();

  for (int kb0 = 0; kb0 < SEQ; kb0 += 64) {
    const int cur = (kb0 >> 6) & 1;
    if (kb0 + 64 < SEQ) stage(cur ^ 1, kb0 + 64);   // overlap copy w/ compute

    // ---- scores: 16 q x 64 keys (4 C-tiles), K-dim = DK = 64 (2 WMMAs each)
    fx8 sc[4];
#pragma unroll
    for (int kt = 0; kt < 4; ++kt) {
      const bf16_t* kr = &ldsK[cur][(kt*16 + l16) * DK + hf*8];
      bfx16 b0 = load_frag(kr);
      bfx16 b1 = load_frag(kr + 32);
      fx8 z = (fx8){};
      z = wmma_bf16(aq0, b0, z);
      z = wmma_bf16(aq1, b1, z);
      sc[kt] = z;
    }
#pragma unroll
    for (int kt = 0; kt < 4; ++kt)
#pragma unroll
      for (int v = 0; v < 8; ++v) sc[kt][v] *= 0.125f;   // 1/sqrt(64)

    // ---- online softmax (row = q). C-layout row m = v + hf*8, col = lane%16.
#pragma unroll
    for (int v = 0; v < 8; ++v) {
      float bm = fmaxf(fmaxf(sc[0][v], sc[1][v]), fmaxf(sc[2][v], sc[3][v]));
#pragma unroll
      for (int off = 1; off < 16; off <<= 1) bm = fmaxf(bm, __shfl_xor(bm, off));
      float mn = fmaxf(mrow[v], bm);
      float al = __expf(mrow[v] - mn);
      mrow[v] = mn;
      lrow[v] *= al;
#pragma unroll
      for (int nt = 0; nt < 4; ++nt) accO[nt][v] *= al;
      float s = 0.f;
#pragma unroll
      for (int kt = 0; kt < 4; ++kt) {
        float p = __expf(sc[kt][v] - mn);
        sc[kt][v] = p;
        s += p;
      }
#pragma unroll
      for (int off = 1; off < 16; off <<= 1) s += __shfl_xor(s, off);
      lrow[v] += s;
    }

    // ---- P (C-layout fp32) -> LDS row-major [q][key] bf16 -> A-layout frags
#pragma unroll
    for (int kt = 0; kt < 4; ++kt)
#pragma unroll
      for (int v = 0; v < 8; ++v)
        lpw[(v + hf*8) * 64 + kt*16 + l16] = (bf16_t)sc[kt][v];

    bfx16 pa0 = load_frag(lpw + l16*64 +  0 + hf*8);   // keys  0..31 of block
    bfx16 pa1 = load_frag(lpw + l16*64 + 32 + hf*8);   // keys 32..63 of block

    // ---- O += P @ V from the shared LDS V tile ([d][key])
#pragma unroll
    for (int nt = 0; nt < 4; ++nt) {
      const bf16_t* vr = &ldsV[cur][(nt*16 + l16) * 64 + hf*8];
      accO[nt] = wmma_bf16(pa0, load_frag(vr), accO[nt]);
      accO[nt] = wmma_bf16(pa1, load_frag(vr + 32), accO[nt]);
    }

    WAIT_ASYNC();        // our async copies into buf^1 are done
    __syncthreads();     // everyone finished reading buf & copying buf^1
  }

  // ---- normalize + write combined heads [NTOK, D_MODEL] bf16
#pragma unroll
  for (int v = 0; v < 8; ++v) {
    float inv = 1.f / lrow[v];
    int r = b * SEQ + q0 + v + hf*8;
#pragma unroll
    for (int nt = 0; nt < 4; ++nt) {
      int c = h * DK + nt*16 + l16;
      attnC[(size_t)r * D_MODEL + c] = (bf16_t)(accO[nt][v] * inv);
    }
  }
}

// ---------------- host launcher ----------------
extern "C" void kernel_launch(void* const* d_in, const int* in_sizes, int n_in,
                              void* d_out, int out_size, void* d_ws, size_t ws_size,
                              hipStream_t stream) {
  (void)in_sizes; (void)n_in; (void)out_size; (void)ws_size;
  const float* Q  = (const float*)d_in[0];
  const float* K  = (const float*)d_in[1];
  const float* V  = (const float*)d_in[2];
  const float* Wq = (const float*)d_in[3];
  const float* bq = (const float*)d_in[4];
  const float* Wk = (const float*)d_in[5];
  const float* bk = (const float*)d_in[6];
  const float* Wv = (const float*)d_in[7];
  const float* bv = (const float*)d_in[8];
  const float* Wo = (const float*)d_in[9];
  const float* bo = (const float*)d_in[10];

  // workspace carve-up (~120 MB total, all bf16)
  char* ws = (char*)d_ws;
  size_t off = 0;
  auto carve = [&](size_t bytes) {
    void* p = ws + off;
    off += (bytes + 255) & ~(size_t)255;
    return p;
  };
  const size_t actBytes = (size_t)NTOK * D_MODEL * 2;      // 16 MB
  const size_t wBytes   = (size_t)D_MODEL * D_MODEL * 2;   //  2 MB
  bf16_t* Qb    = (bf16_t*)carve(actBytes);
  bf16_t* Kb    = (bf16_t*)carve(actBytes);
  bf16_t* Vb    = (bf16_t*)carve(actBytes);
  bf16_t* WqT   = (bf16_t*)carve(wBytes);
  bf16_t* WkT   = (bf16_t*)carve(wBytes);
  bf16_t* WvT   = (bf16_t*)carve(wBytes);
  bf16_t* WoT   = (bf16_t*)carve(wBytes);
  bf16_t* qpj   = (bf16_t*)carve(actBytes);   // [B,H,S,DK]
  bf16_t* kpj   = (bf16_t*)carve(actBytes);   // [B,H,S,DK]
  bf16_t* vTj   = (bf16_t*)carve(actBytes);   // [B,H,DK,S]
  bf16_t* attnC = (bf16_t*)carve(actBytes);   // [NTOK, D_MODEL]

  // 1) fp32 -> bf16 activations
  {
    int n4 = NTOK * D_MODEL / 4;
    int blocks = (n4 + 255) / 256;
    cvt_bf16_4<<<blocks, 256, 0, stream>>>(Q, Qb, n4);
    cvt_bf16_4<<<blocks, 256, 0, stream>>>(K, Kb, n4);
    cvt_bf16_4<<<blocks, 256, 0, stream>>>(V, Vb, n4);
  }
  // 2) weights: fp32 [K,N] -> bf16 [N,K]
  {
    int n = D_MODEL * D_MODEL;
    int blocks = (n + 255) / 256;
    cvt_tr<<<blocks, 256, 0, stream>>>(Wq, WqT, D_MODEL, D_MODEL);
    cvt_tr<<<blocks, 256, 0, stream>>>(Wk, WkT, D_MODEL, D_MODEL);
    cvt_tr<<<blocks, 256, 0, stream>>>(Wv, WvT, D_MODEL, D_MODEL);
    cvt_tr<<<blocks, 256, 0, stream>>>(Wo, WoT, D_MODEL, D_MODEL);
  }
  // 3) projections (WMMA)
  {
    dim3 grid(NTOK / 64, D_MODEL / 64);
    gemm_bf16_k<<<grid, 32, 0, stream>>>(Qb, WqT, bq, qpj, nullptr, NTOK, D_MODEL, D_MODEL, 0);
    gemm_bf16_k<<<grid, 32, 0, stream>>>(Kb, WkT, bk, kpj, nullptr, NTOK, D_MODEL, D_MODEL, 0);
    gemm_bf16_k<<<grid, 32, 0, stream>>>(Vb, WvT, bv, vTj, nullptr, NTOK, D_MODEL, D_MODEL, 1);
  }
  // 4) flash attention (WMMA + async KV staging), 4 q-tiles per block
  {
    dim3 grid(SEQ / 64, BATCH * NHEADS);
    flash_k<<<grid, 128, 0, stream>>>(qpj, kpj, vTj, attnC);
  }
  // 5) output projection -> fp32 d_out
  {
    dim3 grid(NTOK / 64, D_MODEL / 64);
    gemm_bf16_k<<<grid, 32, 0, stream>>>(attnC, WoT, bo, nullptr, (float*)d_out,
                                         NTOK, D_MODEL, D_MODEL, 2);
  }
}
